// RecurrentMoE_84370337562785
// MI455X (gfx1250) — matrix operand order repaired and checked by reading, not compiled, not run on hardware
//
#include <hip/hip_runtime.h>
#include <hip/hip_bf16.h>
#include <math.h>

#define DIM 1024
#define NB  8
#define NT  1024
#define NS  8
#define NE  8
#define NH  8
#define HD  128
#define NKR 4
#define NKW 2
#define NKT 2
#define TKV 1032
#define LNEPS 1e-5f

typedef __attribute__((ext_vector_type(16))) __bf16 v16bf;
typedef __attribute__((ext_vector_type(8)))  float  v8f;
typedef __attribute__((ext_vector_type(2)))  float  v2f;

union BFrag { v16bf v; uint4 q[2]; };
union U4H8  { uint4 q; __bf16 h[8]; };

__device__ inline v8f wmma_bf16(const BFrag& a, const BFrag& b, v8f c) {
  return __builtin_amdgcn_wmma_f32_16x16x32_bf16(false, a.v, false, b.v, (short)0, c, false, false);
}
__device__ inline v8f wmma_f32k4(v2f a, v2f b, v8f c) {
  return __builtin_amdgcn_wmma_f32_16x16x4_f32(false, a, false, b, (short)0, c, false, false);
}

__device__ inline float block_reduce_sum(float v, float* red, int tid) {
  red[tid] = v; __syncthreads();
  for (int off = 128; off > 0; off >>= 1) {
    if (tid < off) red[tid] += red[tid + off];
    __syncthreads();
  }
  float r = red[0]; __syncthreads();
  return r;
}

// ---------------------------------------------------------------------------
// State read + expert gate (tiny): scores, top-KR, gather, LN, gate top-2.
// ---------------------------------------------------------------------------
__global__ __launch_bounds__(256) void k_state_read(
    const float* __restrict__ state0, const float* __restrict__ W_rg,
    const float* __restrict__ g_moe, const float* __restrict__ b_moe,
    const float* __restrict__ W_gate,
    int* __restrict__ read_idx_ws, float* __restrict__ latent_read_ws,
    float* __restrict__ wsum_ws, float* __restrict__ out_idx_experts,
    float* __restrict__ out_read_idx)
{
  const int b = blockIdx.x, tid = threadIdx.x;
  __shared__ float red[256];
  __shared__ float sc[NS];
  __shared__ int   ridx[NKR];
  __shared__ float lnrow[DIM];
  __shared__ float glog[NE];

  for (int s = 0; s < NS; ++s) {
    const float* row = state0 + ((size_t)b * NS + s) * DIM;
    float p = 0.f;
    for (int d = tid; d < DIM; d += 256) p += row[d] * W_rg[d];
    float t = block_reduce_sum(p, red, tid);
    if (tid == 0) sc[s] = t;
  }
  __syncthreads();
  if (tid == 0) {
    bool used[NS]; for (int s = 0; s < NS; ++s) used[s] = false;
    for (int j = 0; j < NKR; ++j) {
      int best = -1;
      for (int s = 0; s < NS; ++s)
        if (!used[s] && (best < 0 || sc[s] > sc[best])) best = s;
      used[best] = true; ridx[j] = best;
      read_idx_ws[b * NKR + j] = best;
      out_read_idx[b * NKR + j] = (float)best;
    }
  }
  __syncthreads();

  for (int j = 0; j < NKR; ++j) {
    const float* row = state0 + ((size_t)b * NS + ridx[j]) * DIM;
    float p = 0.f;
    for (int d = tid; d < DIM; d += 256) {
      float v = row[d];
      latent_read_ws[((size_t)b * NKR + j) * DIM + d] = v;
      p += v;
    }
    float mean = block_reduce_sum(p, red, tid) * (1.f / DIM);
    float vv = 0.f;
    for (int d = tid; d < DIM; d += 256) { float t = row[d] - mean; vv += t * t; }
    float var = block_reduce_sum(vv, red, tid) * (1.f / DIM);
    float rstd = rsqrtf(var + LNEPS);
    for (int d = tid; d < DIM; d += 256)
      lnrow[d] = (row[d] - mean) * rstd * g_moe[d] + b_moe[d];
    __syncthreads();
    for (int e = 0; e < NE; ++e) {
      float q = 0.f;
      for (int d = tid; d < DIM; d += 256) q += lnrow[d] * W_gate[(size_t)e * DIM + d];
      float t = block_reduce_sum(q, red, tid);
      if (tid == 0) glog[e] = t;
    }
    __syncthreads();
    if (tid == 0) {
      int e0 = 0;
      for (int e = 1; e < NE; ++e) if (glog[e] > glog[e0]) e0 = e;
      int e1 = -1;
      for (int e = 0; e < NE; ++e)
        if (e != e0 && (e1 < 0 || glog[e] > glog[e1])) e1 = e;
      float mx = fmaxf(glog[e0], glog[e1]);
      float w0 = expf(glog[e0] - mx), w1 = expf(glog[e1] - mx);
      float ss = w0 + w1; w0 /= ss; w1 /= ss;
      out_idx_experts[(b * NKR + j) * NKT + 0] = (float)e0;
      out_idx_experts[(b * NKR + j) * NKT + 1] = (float)e1;
      wsum_ws[b * NKR + j] = w0 + w1;
    }
    __syncthreads();
  }
}

__global__ void k_wbar(const float* __restrict__ wsum, float* __restrict__ coef) {
  int r = threadIdx.x;
  if (r < NKR) {
    float s = 0.f;
    for (int b = 0; b < NB; ++b) s += wsum[b * NKR + r];
    float wbar = s / (float)(NB * NKT);
    coef[r] = 0.25f * wbar;
  }
}

// ---------------------------------------------------------------------------
// f32 WMMA GEMM:  Out = sgn * X * X^T   (X 1024x1024, symmetric/skew products)
// ---------------------------------------------------------------------------
template <bool NEG>
__global__ __launch_bounds__(256) void k_gemm_f32_xxt(
    const float* __restrict__ X, float* __restrict__ Out)
{
  __shared__ float Als[128 * 36];
  __shared__ float Bls[128 * 36];
  const int tid = threadIdx.x, lane = tid & 31, w = tid >> 5;
  const int wm = w & 3, wn = w >> 2, l16 = lane & 15, hi = lane >> 4;
  const int m0 = blockIdx.x * 128, n0 = blockIdx.y * 128;
  const int srow = tid >> 1, scol = (tid & 1) * 16;

  v8f acc[2][4];
  v8f z8; for (int i = 0; i < 8; ++i) z8[i] = 0.f;
  for (int a = 0; a < 2; ++a) for (int c = 0; c < 4; ++c) acc[a][c] = z8;

  for (int kt = 0; kt < 32; ++kt) {
    const int kb = kt * 32;
    const float* pa = X + (size_t)(m0 + srow) * DIM + kb + scol;
    const float* pb = X + (size_t)(n0 + srow) * DIM + kb + scol;
    for (int i = 0; i < 16; i += 4) {
      *(float4*)&Als[srow * 36 + scol + i] = *(const float4*)(pa + i);
      *(float4*)&Bls[srow * 36 + scol + i] = *(const float4*)(pb + i);
    }
    __syncthreads();
    for (int ks = 0; ks < 8; ++ks) {
      const int kk = ks * 4 + hi * 2;
      v2f af[2], bf[4];
      for (int tm = 0; tm < 2; ++tm)
        af[tm] = *(const v2f*)&Als[(wm * 32 + tm * 16 + l16) * 36 + kk];
      for (int tn = 0; tn < 4; ++tn)
        bf[tn] = *(const v2f*)&Bls[(wn * 64 + tn * 16 + l16) * 36 + kk];
      for (int tm = 0; tm < 2; ++tm)
        for (int tn = 0; tn < 4; ++tn)
          acc[tm][tn] = wmma_f32k4(af[tm], bf[tn], acc[tm][tn]);
    }
    __syncthreads();
  }
  for (int tm = 0; tm < 2; ++tm)
    for (int tn = 0; tn < 4; ++tn) {
      const int nb = m0 + wm * 32 + tm * 16 + 8 * hi;
      const int o  = n0 + wn * 64 + tn * 16 + l16;
      for (int r = 0; r < 8; ++r) {
        float v = acc[tm][tn][r];
        Out[(size_t)(nb + r) * DIM + o] = NEG ? -v : v;
      }
    }
}

// diag extraction: all Ck symmetric -> diag(X*Y) = sum_j X[i,j]*Y[i,j]
__global__ __launch_bounds__(256) void k_diag(
    const float* __restrict__ C2, const float* __restrict__ C4,
    const float* __restrict__ C8, float* __restrict__ dvec)
{
  const int i = blockIdx.x, tid = threadIdx.x;
  __shared__ float red[4 * 256];
  float p[4] = {0.f, 0.f, 0.f, 0.f};
  const float* r2 = C2 + (size_t)i * DIM;
  const float* r4 = C4 + (size_t)i * DIM;
  const float* r8 = C8 + (size_t)i * DIM;
  for (int j = tid; j < DIM; j += 256) {
    float a2 = r2[j], a4 = r4[j], a8 = r8[j];
    p[0] += a2 * a4; p[1] += a2 * a8; p[2] += a4 * a8; p[3] += a8 * a8;
  }
  for (int q = 0; q < 4; ++q) red[q * 256 + tid] = p[q];
  __syncthreads();
  for (int off = 128; off > 0; off >>= 1) {
    if (tid < off)
      for (int q = 0; q < 4; ++q) red[q * 256 + tid] += red[q * 256 + tid + off];
    __syncthreads();
  }
  if (tid == 0) {
    dvec[0 * DIM + i] = r2[i];        // d2
    dvec[1 * DIM + i] = r4[i];        // d4
    dvec[2 * DIM + i] = red[0];       // d6
    dvec[3 * DIM + i] = r8[i];        // d8
    dvec[4 * DIM + i] = red[256];     // d10
    dvec[5 * DIM + i] = red[512];     // d12
    dvec[6 * DIM + i] = red[768];     // d16
  }
}

__global__ void k_qdiag(const float* __restrict__ dvec,
                        const float* __restrict__ coef,
                        float* __restrict__ qdiag)
{
  const int i = blockIdx.x * 256 + threadIdx.x;
  float d2  = dvec[0 * DIM + i], d4  = dvec[1 * DIM + i], d6 = dvec[2 * DIM + i];
  float d8  = dvec[3 * DIM + i], d10 = dvec[4 * DIM + i], d12 = dvec[5 * DIM + i];
  float d16 = dvec[6 * DIM + i];
  for (int r = 0; r < NKR; ++r) {
    float c = coef[r], c2 = c * c;
    float e2 = c2, e4 = e2 * e2, e6 = e4 * e2, e8 = e4 * e4;
    float e10 = e8 * e2, e12 = e8 * e4, e16 = e8 * e8;
    float q = 1.f + 2.f * (e2 * d2 + e4 * d4 + e6 * d6 + e8 * d8 +
                           e10 * d10 + e12 * d12 + e16 * d16);
    qdiag[r * DIM + i] = q;
  }
}

// ---------------------------------------------------------------------------
// Core average + slot write gating + convex state update (tiny).
// ---------------------------------------------------------------------------
__global__ __launch_bounds__(256) void k_core_write(
    const float* __restrict__ state0, const float* __restrict__ latent_read_ws,
    const float* __restrict__ qdiag, const int* __restrict__ read_idx_ws,
    const float* __restrict__ g_slot, const float* __restrict__ b_slot,
    const float* __restrict__ W_slot,
    float* __restrict__ state_dout, float* __restrict__ state_ws,
    float* __restrict__ out_write_idx)
{
  const int b = blockIdx.x, tid = threadIdx.x;
  __shared__ float red[256];
  __shared__ float tc[DIM];
  __shared__ float ss[NS];
  __shared__ float alpha[NS];
  __shared__ float asum_s;

  for (int d = tid; d < DIM; d += 256) {
    float a = 0.f;
    for (int r = 0; r < NKR; ++r)
      a += tanhf(latent_read_ws[((size_t)b * NKR + r) * DIM + d] * qdiag[r * DIM + d]);
    tc[d] = tanhf(a * (1.f / NKR));
  }
  for (int s = 0; s < NS; ++s) {
    const float* row = state0 + ((size_t)b * NS + s) * DIM;
    float p = 0.f;
    for (int d = tid; d < DIM; d += 256) p += row[d];
    float mean = block_reduce_sum(p, red, tid) * (1.f / DIM);
    float vv = 0.f;
    for (int d = tid; d < DIM; d += 256) { float t = row[d] - mean; vv += t * t; }
    float var = block_reduce_sum(vv, red, tid) * (1.f / DIM);
    float rstd = rsqrtf(var + LNEPS);
    float q = 0.f;
    for (int d = tid; d < DIM; d += 256)
      q += ((row[d] - mean) * rstd * g_slot[d] + b_slot[d]) * W_slot[d];
    float t = block_reduce_sum(q, red, tid);
    if (tid == 0) ss[s] = t;
  }
  __syncthreads();
  if (tid == 0) {
    int ridx[NKR]; float lr[NKR];
    for (int j = 0; j < NKR; ++j) { ridx[j] = read_idx_ws[b * NKR + j]; lr[j] = ss[ridx[j]]; }
    int w0 = 0;
    for (int j = 1; j < NKR; ++j) if (lr[j] > lr[w0]) w0 = j;
    int w1 = -1;
    for (int j = 0; j < NKR; ++j)
      if (j != w0 && (w1 < 0 || lr[j] > lr[w1])) w1 = j;
    const int wi0 = ridx[w0], wi1 = ridx[w1];
    out_write_idx[b * NKW + 0] = (float)wi0;
    out_write_idx[b * NKW + 1] = (float)wi1;
    // note: softmax over the FIRST KW columns of logits_read (as in source)
    float mx = fmaxf(lr[0], lr[1]);
    float e0 = expf(lr[0] - mx), e1 = expf(lr[1] - mx);
    float sm = e0 + e1;
    for (int s = 0; s < NS; ++s) alpha[s] = 0.f;
    alpha[wi0] += e0 / sm; alpha[wi1] += e1 / sm;
    float as = 0.f; for (int s = 0; s < NS; ++s) as += alpha[s];
    asum_s = as;
  }
  __syncthreads();
  const float om = 1.f - asum_s;
  for (int s = 0; s < NS; ++s)
    for (int d = tid; d < DIM; d += 256) {
      float v = alpha[s] * tc[d] + om * state0[((size_t)b * NS + s) * DIM + d];
      state_dout[((size_t)b * NS + s) * DIM + d] = v;
      state_ws [((size_t)b * NS + s) * DIM + d] = v;
    }
}

__global__ void k_pe(float* __restrict__ pe) {
  const int idx = blockIdx.x * 256 + threadIdx.x;
  const int t = idx >> 10, d = idx & 1023;
  const int half = DIM / 2;
  const int di = (d < half) ? d : d - half;
  const float fr = powf((float)NS, (float)di / (float)half);
  pe[idx] = (d < half) ? sinf((float)t / fr) : cosf((float)t / fr);
}

// ---------------------------------------------------------------------------
// bf16 WMMA GEMM:  Out[n,o] = epilogue( sum_k A[n,k] * W[o,k] + bias[o] )
// block 128x128, 8 waves (4 M x 2 N), wave tile 32x64, K step 32.
// ---------------------------------------------------------------------------
enum { EP_PE = 0, EP_BIAS_BF16 = 1, EP_RES_F32 = 2, EP_GELU_BF16 = 3 };

template <bool A_BF16, int EP>
__global__ __launch_bounds__(256) void k_gemm(
    const void* __restrict__ Aptr, const float* __restrict__ W,
    const float* __restrict__ bias, const float* __restrict__ extra,
    void* __restrict__ Out, int Mrows)
{
  __shared__ __bf16 Als[128 * 40];
  __shared__ __bf16 Bls[128 * 40];
  const int tid = threadIdx.x, lane = tid & 31, w = tid >> 5;
  const int wm = w & 3, wn = w >> 2, l16 = lane & 15, hi = lane >> 4;
  const int m0 = blockIdx.x * 128, n0 = blockIdx.y * 128;
  const int srow = tid >> 1, scol = (tid & 1) * 16;

  v8f acc[2][4];
  v8f z8; for (int i = 0; i < 8; ++i) z8[i] = 0.f;
  for (int a = 0; a < 2; ++a) for (int c = 0; c < 4; ++c) acc[a][c] = z8;

  for (int kt = 0; kt < 32; ++kt) {
    const int kb = kt * 32;
    // ---- stage A (convert f32 -> bf16 if needed) ----
    {
      const int gr = m0 + srow;
      uint4 u0 = {0, 0, 0, 0}, u1 = {0, 0, 0, 0};
      if (gr < Mrows) {
        if (A_BF16) {
          const __bf16* As = (const __bf16*)Aptr + (size_t)gr * DIM + kb + scol;
          u0 = *(const uint4*)(As);
          u1 = *(const uint4*)(As + 8);
        } else {
          const float* As = (const float*)Aptr + (size_t)gr * DIM + kb + scol;
          float tmp[16];
          *(float4*)&tmp[0]  = *(const float4*)(As + 0);
          *(float4*)&tmp[4]  = *(const float4*)(As + 4);
          *(float4*)&tmp[8]  = *(const float4*)(As + 8);
          *(float4*)&tmp[12] = *(const float4*)(As + 12);
          U4H8 a0, a1;
          for (int i = 0; i < 8; ++i) { a0.h[i] = (__bf16)tmp[i]; a1.h[i] = (__bf16)tmp[8 + i]; }
          u0 = a0.q; u1 = a1.q;
        }
      }
      *(uint4*)&Als[srow * 40 + scol]     = u0;
      *(uint4*)&Als[srow * 40 + scol + 8] = u1;
    }
    // ---- stage B (weight rows o, K-contiguous; N=1024 exact) ----
    {
      const float* Ws = W + (size_t)(n0 + srow) * DIM + kb + scol;
      float tmp[16];
      *(float4*)&tmp[0]  = *(const float4*)(Ws + 0);
      *(float4*)&tmp[4]  = *(const float4*)(Ws + 4);
      *(float4*)&tmp[8]  = *(const float4*)(Ws + 8);
      *(float4*)&tmp[12] = *(const float4*)(Ws + 12);
      U4H8 b0, b1;
      for (int i = 0; i < 8; ++i) { b0.h[i] = (__bf16)tmp[i]; b1.h[i] = (__bf16)tmp[8 + i]; }
      *(uint4*)&Bls[srow * 40 + scol]     = b0.q;
      *(uint4*)&Bls[srow * 40 + scol + 8] = b1.q;
    }
    __syncthreads();
    // ---- fragments (ISA 16-bit A/B layouts) ----
    BFrag afrag[2], bfrag[4];
    for (int tm = 0; tm < 2; ++tm) {
      const int row = wm * 32 + tm * 16 + l16;
      const int c0 = hi ? 8 : 0;
      afrag[tm].q[0] = *(const uint4*)&Als[row * 40 + c0];
      afrag[tm].q[1] = *(const uint4*)&Als[row * 40 + c0 + 16];
    }
    for (int tn = 0; tn < 4; ++tn) {
      const int col = wn * 64 + tn * 16 + l16;
      const int c0 = hi ? 8 : 0;
      bfrag[tn].q[0] = *(const uint4*)&Bls[col * 40 + c0];
      bfrag[tn].q[1] = *(const uint4*)&Bls[col * 40 + c0 + 16];
    }
    for (int tm = 0; tm < 2; ++tm)
      for (int tn = 0; tn < 4; ++tn)
        acc[tm][tn] = wmma_bf16(afrag[tm], bfrag[tn], acc[tm][tn]);
    __syncthreads();
  }
  // ---- epilogue ----
  for (int tm = 0; tm < 2; ++tm)
    for (int tn = 0; tn < 4; ++tn) {
      const int nb = m0 + wm * 32 + tm * 16 + 8 * hi;
      const int o  = n0 + wn * 64 + tn * 16 + l16;
      const float bo = bias[o];
      for (int r = 0; r < 8; ++r) {
        const int n = nb + r;
        if (n >= Mrows) continue;
        float v = acc[tm][tn][r] + bo;
        if (EP == EP_PE) {
          v += extra[(size_t)(n & (NT - 1)) * DIM + o];
          ((float*)Out)[(size_t)n * DIM + o] = v;
        } else if (EP == EP_BIAS_BF16) {
          ((__bf16*)Out)[(size_t)n * DIM + o] = (__bf16)v;
        } else if (EP == EP_RES_F32) {
          v += extra[(size_t)n * DIM + o];
          ((float*)Out)[(size_t)n * DIM + o] = v;
        } else { // EP_GELU_BF16 (exact erf gelu)
          float g = 0.5f * v * (1.f + erff(v * 0.70710678118654752f));
          ((__bf16*)Out)[(size_t)n * DIM + o] = (__bf16)g;
        }
      }
    }
}

// ---------------------------------------------------------------------------
// LayerNorm (f32 in -> bf16 out), one block per row.
// ---------------------------------------------------------------------------
__global__ __launch_bounds__(256) void k_ln_bf16(
    const float* __restrict__ src, __bf16* __restrict__ dst,
    const float* __restrict__ g, const float* __restrict__ bb)
{
  __shared__ float red[256];
  const int tid = threadIdx.x;
  const size_t r = blockIdx.x;
  const float* row = src + r * DIM;
  float s = 0.f;
  for (int d = tid; d < DIM; d += 256) s += row[d];
  const float mean = block_reduce_sum(s, red, tid) * (1.f / DIM);
  float vv = 0.f;
  for (int d = tid; d < DIM; d += 256) { float t = row[d] - mean; vv += t * t; }
  const float var = block_reduce_sum(vv, red, tid) * (1.f / DIM);
  const float rstd = rsqrtf(var + LNEPS);
  for (int d = tid; d < DIM; d += 256)
    dst[r * DIM + d] = (__bf16)((row[d] - mean) * rstd * g[d] + bb[d]);
}

// kv = LN(concat(latent0, state)) with row splice
__global__ __launch_bounds__(256) void k_ln_kv(
    const float* __restrict__ latent0, const float* __restrict__ state_ws,
    __bf16* __restrict__ dst, const float* __restrict__ g,
    const float* __restrict__ bb)
{
  __shared__ float red[256];
  const int tid = threadIdx.x;
  const size_t rw = blockIdx.x;
  const int b = (int)(rw / TKV), t = (int)(rw % TKV);
  const float* row = (t < NT)
      ? latent0 + ((size_t)b * NT + t) * DIM
      : state_ws + ((size_t)b * NS + (t - NT)) * DIM;
  float s = 0.f;
  for (int d = tid; d < DIM; d += 256) s += row[d];
  const float mean = block_reduce_sum(s, red, tid) * (1.f / DIM);
  float vv = 0.f;
  for (int d = tid; d < DIM; d += 256) { float t2 = row[d] - mean; vv += t2 * t2; }
  const float var = block_reduce_sum(vv, red, tid) * (1.f / DIM);
  const float rstd = rsqrtf(var + LNEPS);
  for (int d = tid; d < DIM; d += 256)
    dst[rw * DIM + d] = (__bf16)((row[d] - mean) * rstd * g[d] + bb[d]);
}

// ---------------------------------------------------------------------------
// Flash attention: per (b,h,qblock of 128). 8 waves x 16 query rows.
// ---------------------------------------------------------------------------
__global__ __launch_bounds__(256) void k_attn(
    const __bf16* __restrict__ qp, const __bf16* __restrict__ kp,
    const __bf16* __restrict__ vp, __bf16* __restrict__ o)
{
  const int qb = blockIdx.x, h = blockIdx.y, b = blockIdx.z;
  __shared__ __bf16 Qls[128 * 136];
  __shared__ __bf16 Kls[32 * 136];
  __shared__ __bf16 Vtls[128 * 40];
  __shared__ __bf16 Pls[8 * 16 * 40];
  const int tid = threadIdx.x, lane = tid & 31, w = tid >> 5;
  const int l16 = lane & 15, hi = lane >> 4;

  const size_t qbase  = ((size_t)b * NT + qb * 128) * DIM + h * HD;
  const size_t kvbase = (size_t)b * TKV * DIM + h * HD;

  for (int u = tid; u < 2048; u += 256) {
    const int row = u >> 4, c8 = u & 15;
    uint4 val = *(const uint4*)(qp + qbase + (size_t)row * DIM + c8 * 8);
    *(uint4*)&Qls[row * 136 + c8 * 8] = val;
  }
  __syncthreads();

  BFrag qf[4];
  for (int ks = 0; ks < 4; ++ks) {
    const int row = 16 * w + l16;
    const int c0 = ks * 32 + (hi ? 8 : 0);
    qf[ks].q[0] = *(const uint4*)&Qls[row * 136 + c0];
    qf[ks].q[1] = *(const uint4*)&Qls[row * 136 + c0 + 16];
  }

  v8f Of[8];
  v8f z8; for (int i = 0; i < 8; ++i) z8[i] = 0.f;
  for (int f = 0; f < 8; ++f) Of[f] = z8;
  float mrow[8], lrow[8];
  for (int r = 0; r < 8; ++r) { mrow[r] = -1e30f; lrow[r] = 0.f; }
  const float scale = 0.088388347648318447f; // 1/sqrt(128)

  for (int kt = 0; kt < 33; ++kt) {
    const int tok0 = kt * 32;
    __syncthreads();
    // stage K tile [tok][d] bf16
    for (int u = tid; u < 512; u += 256) {
      const int tr = u >> 4, c8 = u & 15;
      uint4 val = {0, 0, 0, 0};
      if (tok0 + tr < TKV)
        val = *(const uint4*)(kp + kvbase + (size_t)(tok0 + tr) * DIM + c8 * 8);
      *(uint4*)&Kls[tr * 136 + c8 * 8] = val;
    }
    // stage V transposed: Vt[d][tok]
    for (int u = tid; u < 512; u += 256) {
      const int tr = u >> 4, c8 = u & 15;
      U4H8 val; val.q = uint4{0, 0, 0, 0};
      if (tok0 + tr < TKV)
        val.q = *(const uint4*)(vp + kvbase + (size_t)(tok0 + tr) * DIM + c8 * 8);
      for (int i = 0; i < 8; ++i) Vtls[(c8 * 8 + i) * 40 + tr] = val.h[i];
    }
    __syncthreads();

    // scores: S = Q K^T for 16 rows x 32 keys (2 tiles x 4 k-steps)
    v8f sf[2] = {z8, z8};
    for (int tn = 0; tn < 2; ++tn)
      for (int ks = 0; ks < 4; ++ks) {
        const int tokr = tn * 16 + l16;
        const int c0 = ks * 32 + (hi ? 8 : 0);
        BFrag kf;
        kf.q[0] = *(const uint4*)&Kls[tokr * 136 + c0];
        kf.q[1] = *(const uint4*)&Kls[tokr * 136 + c0 + 16];
        sf[tn] = wmma_bf16(qf[ks], kf, sf[tn]);
      }
    for (int tn = 0; tn < 2; ++tn) {
      const int colg = tok0 + tn * 16 + l16;
      if (colg >= TKV) { for (int r = 0; r < 8; ++r) sf[tn][r] = -1e30f; }
      else             { for (int r = 0; r < 8; ++r) sf[tn][r] *= scale; }
    }
    // online softmax (row stats reduced over the 16 lanes of each half)
    float corr[8];
    for (int r = 0; r < 8; ++r) {
      float mt = fmaxf(sf[0][r], sf[1][r]);
      for (int off = 1; off < 16; off <<= 1) mt = fmaxf(mt, __shfl_xor(mt, off, 32));
      const float mn = fmaxf(mrow[r], mt);
      const float c = __expf(mrow[r] - mn);
      const float p0 = __expf(sf[0][r] - mn);
      const float p1 = __expf(sf[1][r] - mn);
      sf[0][r] = p0; sf[1][r] = p1;
      float rs = p0 + p1;
      for (int off = 1; off < 16; off <<= 1) rs += __shfl_xor(rs, off, 32);
      lrow[r] = lrow[r] * c + rs;
      mrow[r] = mn; corr[r] = c;
    }
    for (int f = 0; f < 8; ++f)
      for (int r = 0; r < 8; ++r) Of[f][r] *= corr[r];
    // relayout P (C-frag) -> A-frag via per-wave LDS scratch
    for (int tn = 0; tn < 2; ++tn)
      for (int r = 0; r < 8; ++r) {
        const int M = r + 8 * hi;
        Pls[w * 16 * 40 + M * 40 + tn * 16 + l16] = (__bf16)sf[tn][r];
      }
    asm volatile("s_wait_dscnt 0" ::: "memory");
    BFrag pf;
    {
      const int row = l16;
      const int c0 = hi ? 8 : 0;
      pf.q[0] = *(const uint4*)&Pls[w * 16 * 40 + row * 40 + c0];
      pf.q[1] = *(const uint4*)&Pls[w * 16 * 40 + row * 40 + c0 + 16];
    }
    for (int f = 0; f < 8; ++f) {
      const int d = f * 16 + l16;
      const int c0 = hi ? 8 : 0;
      BFrag vf;
      vf.q[0] = *(const uint4*)&Vtls[d * 40 + c0];
      vf.q[1] = *(const uint4*)&Vtls[d * 40 + c0 + 16];
      Of[f] = wmma_bf16(pf, vf, Of[f]);
    }
  }
  for (int f = 0; f < 8; ++f)
    for (int r = 0; r < 8; ++r) {
      const int qrow = qb * 128 + 16 * w + r + 8 * hi;
      const float v = Of[f][r] / lrow[r];
      o[((size_t)b * NT + qrow) * DIM + h * HD + f * 16 + l16] = (__bf16)v;
    }
}

// y = latent2[:, -1] @ W_out^T + b_out  (small, fp32)
__global__ __launch_bounds__(256) void k_final_y(
    const float* __restrict__ latent2, const float* __restrict__ Wout,
    const float* __restrict__ bout, float* __restrict__ y)
{
  const int idx = blockIdx.x * 256 + threadIdx.x;
  const int b = idx >> 10, o = idx & 1023;
  const float* xr = latent2 + ((size_t)(b * NT + (NT - 1))) * DIM;
  const float* wr = Wout + (size_t)o * DIM;
  float acc = 0.f;
  for (int k = 0; k < DIM; k += 4) {
    float4 xa = *(const float4*)(xr + k);
    float4 wa = *(const float4*)(wr + k);
    acc += xa.x * wa.x + xa.y * wa.y + xa.z * wa.z + xa.w * wa.w;
  }
  y[idx] = acc + bout[o];
}

// ---------------------------------------------------------------------------
extern "C" void kernel_launch(void* const* d_in, const int* in_sizes, int n_in,
                              void* d_out, int out_size, void* d_ws, size_t ws_size,
                              hipStream_t stream) {
  (void)in_sizes; (void)n_in; (void)out_size; (void)ws_size;
  const float* x        = (const float*)d_in[0];
  const float* state0   = (const float*)d_in[1];
  const float* W_rg     = (const float*)d_in[2];
  const float* ln_moe_g = (const float*)d_in[3];
  const float* ln_moe_b = (const float*)d_in[4];
  const float* W_gate   = (const float*)d_in[5];
  const float* A_skew   = (const float*)d_in[6];
  const float* ln_slot_g= (const float*)d_in[7];
  const float* ln_slot_b= (const float*)d_in[8];
  const float* W_slot   = (const float*)d_in[9];
  const float* W_oe     = (const float*)d_in[10];
  const float* b_oe     = (const float*)d_in[11];
  const float* ln_q_g   = (const float*)d_in[12];
  const float* ln_q_b   = (const float*)d_in[13];
  const float* ln_kv_g  = (const float*)d_in[14];
  const float* ln_kv_b  = (const float*)d_in[15];
  const float* in_w     = (const float*)d_in[16];
  const float* in_b     = (const float*)d_in[17];
  const float* out_w    = (const float*)d_in[18];
  const float* out_b    = (const float*)d_in[19];
  const float* ln_ffn_g = (const float*)d_in[20];
  const float* ln_ffn_b = (const float*)d_in[21];
  const float* W_fc1    = (const float*)d_in[22];
  const float* b_fc1    = (const float*)d_in[23];
  const float* W_fc2    = (const float*)d_in[24];
  const float* b_fc2    = (const float*)d_in[25];
  const float* W_out    = (const float*)d_in[26];
  const float* b_out    = (const float*)d_in[27];

  float* dout = (float*)d_out;
  float* out_y     = dout;                           // 8192
  float* out_state = dout + 8192;                    // 65536
  float* out_iexp  = dout + 8192 + 65536;            // 64
  float* out_ridx  = out_iexp + 64;                  // 32
  float* out_widx  = out_ridx + 32;                  // 16

  char* wsb = (char*)d_ws;
  size_t cur = 0;
  auto alloc = [&](size_t bytes) -> void* {
    void* p = (void*)(wsb + cur);
    cur += (bytes + 255) & ~(size_t)255;
    return p;
  };
  const size_t M1 = (size_t)NB * NT;    // 8192
  const size_t M2 = (size_t)NB * TKV;   // 8256
  float*  latent0 = (float*) alloc(M1 * DIM * 4);   // latent0/1/2 (reused in place)
  __bf16* qln     = (__bf16*)alloc(M1 * DIM * 2);   // also ln_ffn
  __bf16* kvln    = (__bf16*)alloc(M2 * DIM * 2);   // also h (gelu)
  __bf16* qpb     = (__bf16*)alloc(M1 * DIM * 2);
  __bf16* kpb     = (__bf16*)alloc(M2 * DIM * 2);
  __bf16* vpb     = (__bf16*)alloc(M2 * DIM * 2);
  __bf16* ob      = (__bf16*)alloc(M1 * DIM * 2);
  float*  C2      = (float*) alloc((size_t)DIM * DIM * 4);
  float*  C4      = (float*) alloc((size_t)DIM * DIM * 4);
  float*  C8      = (float*) alloc((size_t)DIM * DIM * 4);
  float*  pe      = (float*) alloc((size_t)NT * DIM * 4);
  float*  lread   = (float*) alloc((size_t)NB * NKR * DIM * 4);
  float*  state_ws= (float*) alloc((size_t)NB * NS * DIM * 4);
  int*    ridx_ws = (int*)   alloc((size_t)NB * NKR * 4);
  float*  wsum    = (float*) alloc((size_t)NB * NKR * 4);
  float*  coef    = (float*) alloc((size_t)NKR * 4);
  float*  dvec    = (float*) alloc((size_t)7 * DIM * 4);
  float*  qdiag   = (float*) alloc((size_t)NKR * DIM * 4);

  // ---- state read + gate ----
  k_state_read<<<NB, 256, 0, stream>>>(state0, W_rg, ln_moe_g, ln_moe_b, W_gate,
                                       ridx_ws, lread, wsum, out_iexp, out_ridx);
  k_wbar<<<1, 64, 0, stream>>>(wsum, coef);
  // ---- Cayley diagonal via even-power Neumann series (f32 WMMA) ----
  k_gemm_f32_xxt<true ><<<dim3(8, 8), 256, 0, stream>>>(A_skew, C2); // C2 = A^2 = -(A A^T)
  k_gemm_f32_xxt<false><<<dim3(8, 8), 256, 0, stream>>>(C2, C4);     // C4 = C2 C2^T (sym)
  k_gemm_f32_xxt<false><<<dim3(8, 8), 256, 0, stream>>>(C4, C8);     // C8 = C4 C4^T (sym)
  k_diag<<<DIM, 256, 0, stream>>>(C2, C4, C8, dvec);
  k_qdiag<<<DIM / 256, 256, 0, stream>>>(dvec, coef, qdiag);
  // ---- core + write gating + state update ----
  k_core_write<<<NB, 256, 0, stream>>>(state0, lread, qdiag, ridx_ws,
                                       ln_slot_g, ln_slot_b, W_slot,
                                       out_state, state_ws, out_widx);
  // ---- positional embedding ----
  k_pe<<<(NT * DIM) / 256, 256, 0, stream>>>(pe);
  // ---- latent0 = x @ W_oe^T + b_oe + pe ----
  k_gemm<false, EP_PE><<<dim3(64, 8), 256, 0, stream>>>(x, W_oe, b_oe, pe, latent0, (int)M1);
  // ---- LN q / kv ----
  k_ln_bf16<<<(int)M1, 256, 0, stream>>>(latent0, qln, ln_q_g, ln_q_b);
  k_ln_kv  <<<(int)M2, 256, 0, stream>>>(latent0, state_ws, kvln, ln_kv_g, ln_kv_b);
  // ---- q/k/v projections (bf16 out) ----
  k_gemm<true, EP_BIAS_BF16><<<dim3(64, 8), 256, 0, stream>>>(qln, in_w, in_b, nullptr, qpb, (int)M1);
  k_gemm<true, EP_BIAS_BF16><<<dim3(65, 8), 256, 0, stream>>>(kvln, in_w + (size_t)DIM * DIM,
                                                              in_b + DIM, nullptr, kpb, (int)M2);
  k_gemm<true, EP_BIAS_BF16><<<dim3(65, 8), 256, 0, stream>>>(kvln, in_w + (size_t)2 * DIM * DIM,
                                                              in_b + 2 * DIM, nullptr, vpb, (int)M2);
  // ---- attention ----
  k_attn<<<dim3(NT / 128, NH, NB), 256, 0, stream>>>(qpb, kpb, vpb, ob);
  // ---- out_proj + residual (latent1 overwrites latent0) ----
  k_gemm<true, EP_RES_F32><<<dim3(64, 8), 256, 0, stream>>>(ob, out_w, out_b, latent0, latent0, (int)M1);
  // ---- FFN ----
  k_ln_bf16<<<(int)M1, 256, 0, stream>>>(latent0, qln, ln_ffn_g, ln_ffn_b);
  k_gemm<true, EP_GELU_BF16><<<dim3(64, 8), 256, 0, stream>>>(qln, W_fc1, b_fc1, nullptr, kvln, (int)M1);
  k_gemm<true, EP_RES_F32><<<dim3(64, 8), 256, 0, stream>>>(kvln, W_fc2, b_fc2, latent0, latent0, (int)M1);
  // ---- final projection of last token ----
  k_final_y<<<(NB * DIM) / 256, 256, 0, stream>>>(latent0, W_out, b_out, out_y);
}
